// Nei_p_15633680958327
// MI455X (gfx1250) — matrix operand order repaired and checked by reading, not compiled, not run on hardware
//
#include <hip/hip_runtime.h>
#include <math.h>

// ---------------- problem constants ----------------
#define NB   4096          // N
#define MM   33            // M (1 target + 32 neighbors)
#define WW   7             // W timesteps
#define FIN  256           // F_IN
#define REP  128
#define G3   384           // 3*REP
#define NW   (NB*WW)       // 28672 flattened (n,w) rows
#define HYPD 128

typedef __attribute__((ext_vector_type(16))) _Float16 v16h;
typedef __attribute__((ext_vector_type(8)))  _Float16 v8h;
typedef __attribute__((ext_vector_type(8)))  float    v8f;

// ---------------- small utility kernels ----------------
__global__ __launch_bounds__(256) void k_cvt_f16(const float* __restrict__ s,
                                                 _Float16* __restrict__ d, int n) {
    int i = blockIdx.x * blockDim.x + threadIdx.x;
    if (i < n) d[i] = (_Float16)s[i];
}

__global__ __launch_bounds__(256) void k_zero_f32(float* p, int n) {
    int i = blockIdx.x * blockDim.x + threadIdx.x;
    if (i < n) p[i] = 0.0f;
}

__global__ __launch_bounds__(256) void k_zero_f16(_Float16* p, int n) {
    int i = blockIdx.x * blockDim.x + threadIdx.x;
    if (i < n) p[i] = (_Float16)0.0f;
}

// ---------------- fused neighbor reduce (the one mandatory pass over X) -----
// For each (n,w):  Xt16[b,f]  = X[n,0,w,f]                       (target, f16)
//                  Xw16[b,f]  = sum_{m=1..32} C[n,m,w]*X[n,m,w,f] (f16)
//                  Psum[b]    = sum_{m=1..32} C[n,m,w]
__global__ __launch_bounds__(256) void k_reduce_neighbors(
    const float* __restrict__ X, const float* __restrict__ C,
    _Float16* __restrict__ Xw16, _Float16* __restrict__ Xt16,
    float* __restrict__ Psum)
{
    __shared__ float cw[MM];
    const int b = blockIdx.x;          // 0..NW-1
    const int n = b / WW;
    const int w = b % WW;
    const int f = threadIdx.x;         // 0..255

    if (f < MM) cw[f] = C[((size_t)n * MM + f) * WW + w];
    __syncthreads();

    const size_t base = (((size_t)n * MM) * WW + w) * FIN + f; // m=0 element
    const size_t strm = (size_t)WW * FIN;                      // stride over m

    // target (m = 0)
    Xt16[(size_t)b * FIN + f] = (_Float16)X[base];

    float acc = 0.0f;
    #pragma unroll 4
    for (int m = 1; m < MM; ++m) {
        if (m + 2 < MM) __builtin_prefetch(X + base + (size_t)(m + 2) * strm, 0, 0);
        acc = fmaf(cw[m], X[base + (size_t)m * strm], acc);
    }
    Xw16[(size_t)b * FIN + f] = (_Float16)acc;

    if (f == 0) {
        float ps = 0.0f;
        for (int m = 1; m < MM; ++m) ps += cw[m];
        Psum[b] = ps;
    }
}

// ---------------- generic WMMA GEMM:  Out = (beta?Out:0) + A @ W^T + rowScale*bias
// A   : (rows, K) f16, row stride lda
// Wt  : (Nout, K) f16 row-major (so B(k,n) = Wt[n*ldw + k])
// Out : (rows, Nout) f32, row stride ldo ; optional f16 mirror Out16 (ldo16)
// act : 0 = none, 1 = tanh
// grid = (rows/16, Nout/64), block = 128 (4 waves, each does a 16x16 tile)
__global__ __launch_bounds__(128) void k_gemm_wmma_f16(
    const _Float16* __restrict__ A, int lda,
    const _Float16* __restrict__ Wt, int ldw,
    float* __restrict__ Out, int ldo,
    _Float16* __restrict__ Out16, int ldo16,
    const float* __restrict__ bias,
    const float* __restrict__ rowScale,
    int K, int beta, int act)
{
    const int lane = threadIdx.x & 31;
    const int wave = threadIdx.x >> 5;
    const int grp  = lane >> 4;        // lane group 0/1
    const int l16  = lane & 15;

    const int row0 = blockIdx.x * 16;
    const int col0 = blockIdx.y * 64 + wave * 16;

    v8f c = {};
    if (beta) {
        #pragma unroll
        for (int v = 0; v < 8; ++v)
            c[v] = Out[(size_t)(row0 + v + 8 * grp) * ldo + col0 + l16];
    }

    // A fragment (16x32 f16): lane-group 0 holds K {0..7,16..23}, group 1 {8..15,24..31}
    const _Float16* pa = A  + (size_t)(row0 + l16) * lda + 8 * grp;
    // B fragment (32x16 f16): lane-group 0 holds K {0..7,16..23}, group 1 {8..15,24..31}; col = l16
    const _Float16* pb = Wt + (size_t)(col0 + l16) * ldw + 8 * grp;

    for (int k0 = 0; k0 < K; k0 += 32) {
        union { v16h v; v8h h[2]; } a, bfr;
        a.h[0]   = *(const v8h*)(pa + k0);        // K = k0+8g+0..7
        a.h[1]   = *(const v8h*)(pa + k0 + 16);   // K = k0+8g+16..23
        bfr.h[0] = *(const v8h*)(pb + k0);        // K = k0+8g+0..7
        bfr.h[1] = *(const v8h*)(pb + k0 + 16);   // K = k0+8g+16..23
        c = __builtin_amdgcn_wmma_f32_16x16x32_f16(
                false, a.v, false, bfr.v, (short)0, c, false, false);
    }

    const float bn = bias ? bias[col0 + l16] : 0.0f;
    #pragma unroll
    for (int v = 0; v < 8; ++v) {
        const int r   = row0 + v + 8 * grp;
        float val = c[v] + (rowScale ? rowScale[r] : 1.0f) * bn;
        if (act == 1) val = tanhf(val);
        Out[(size_t)r * ldo + col0 + l16] = val;
        if (Out16) Out16[(size_t)r * ldo16 + col0 + l16] = (_Float16)val;
    }
}

// ---------------- GRU gate elementwise ----------------
// xg row n at xg + n*xg_stride (pre-offset for timestep); hg (nrows,384);
// hprev (nrows,128). Writes hout/hout16 and optional ys16 (pre-offset, stride).
__global__ __launch_bounds__(256) void k_gru_gates(
    const float* __restrict__ xg, int xg_stride,
    const float* __restrict__ hg,
    const float* __restrict__ hprev,
    float* __restrict__ hout, _Float16* __restrict__ hout16,
    _Float16* __restrict__ ys16, int ys_stride, int nrows)
{
    int i = blockIdx.x * blockDim.x + threadIdx.x;
    if (i >= nrows * REP) return;
    const int n = i >> 7, f = i & (REP - 1);
    const float* x = xg + (size_t)n * xg_stride;
    const float* g = hg + (size_t)n * G3;
    const float xr = x[f], xz = x[f + REP], xn = x[f + 2 * REP];
    const float hr = g[f], hz = g[f + REP], hn = g[f + 2 * REP];
    const float r  = 1.0f / (1.0f + __expf(-(xr + hr)));
    const float z  = 1.0f / (1.0f + __expf(-(xz + hz)));
    const float nn = tanhf(xn + r * hn);
    const float hp = hprev[i];
    const float h  = (1.0f - z) * nn + z * hp;
    hout[i]   = h;
    hout16[i] = (_Float16)h;
    if (ys16) ys16[(size_t)n * ys_stride + f] = (_Float16)h;
}

// ---------------- head: BN + ReLU + out-proj + sigmoid + BCE loss ----------
__global__ __launch_bounds__(128) void k_head(
    const float* __restrict__ h0, const float* __restrict__ Y,
    const float* __restrict__ gamma, const float* __restrict__ bbeta,
    const float* __restrict__ mean,  const float* __restrict__ var,
    const float* __restrict__ outW,  const float* __restrict__ outb,
    float* __restrict__ d_out)
{
    __shared__ float red[HYPD];
    const int n = blockIdx.x, f = threadIdx.x;
    float v = h0[(size_t)n * HYPD + f];
    const float rs = rsqrtf(var[f] + 1e-5f);
    v = (v - mean[f]) * rs * gamma[f] + bbeta[f];
    v = fmaxf(v, 0.0f);
    red[f] = v * outW[f];
    __syncthreads();
    for (int s = HYPD / 2; s > 0; s >>= 1) {
        if (f < s) red[f] += red[f + s];
        __syncthreads();
    }
    if (f == 0) {
        const float y = red[0] + outb[0];
        d_out[1 + n] = 1.0f / (1.0f + __expf(-y));
        const float ty = Y[(size_t)n * MM];   // Y[n,0]
        const float loss = fmaxf(y, 0.0f) - y * ty + log1pf(__expf(-fabsf(y)));
        atomicAdd(d_out, loss * (1.0f / (float)NB));
    }
}

// ---------------- host-side orchestration ----------------
static inline void launch_gemm(hipStream_t s,
    const _Float16* A, int lda, const _Float16* Wt, int ldw,
    float* Out, int ldo, _Float16* Out16, int ldo16,
    const float* bias, const float* rowScale,
    int rows, int K, int Nout, int beta, int act)
{
    dim3 g(rows / 16, Nout / 64), b(128);
    hipLaunchKernelGGL(k_gemm_wmma_f16, g, b, 0, s,
                       A, lda, Wt, ldw, Out, ldo, Out16, ldo16,
                       bias, rowScale, K, beta, act);
}

extern "C" void kernel_launch(void* const* d_in, const int* in_sizes, int n_in,
                              void* d_out, int out_size, void* d_ws, size_t ws_size,
                              hipStream_t stream)
{
    (void)in_sizes; (void)n_in; (void)out_size; (void)ws_size;

    const float* X        = (const float*)d_in[0];
    const float* Y        = (const float*)d_in[1];
    const float* C        = (const float*)d_in[2];
    const float* msg_W    = (const float*)d_in[3];
    const float* msg_b    = (const float*)d_in[4];
    const float* g0_Wih   = (const float*)d_in[5];
    const float* g0_Whh   = (const float*)d_in[6];
    const float* g0_bih   = (const float*)d_in[7];
    const float* g0_bhh   = (const float*)d_in[8];
    const float* g1_Wih   = (const float*)d_in[9];
    const float* g1_Whh   = (const float*)d_in[10];
    const float* g1_bih   = (const float*)d_in[11];
    const float* g1_bhh   = (const float*)d_in[12];
    const float* lin_W    = (const float*)d_in[13];
    const float* lin_b    = (const float*)d_in[14];
    const float* lin2_W   = (const float*)d_in[15];
    const float* lin2_b   = (const float*)d_in[16];
    const float* c_Wih    = (const float*)d_in[17];
    const float* c_Whh    = (const float*)d_in[18];
    const float* c_bih    = (const float*)d_in[19];
    const float* c_bhh    = (const float*)d_in[20];
    const float* hyp_W    = (const float*)d_in[21];
    const float* hyp_b    = (const float*)d_in[22];
    const float* bn_gamma = (const float*)d_in[23];
    const float* bn_beta  = (const float*)d_in[24];
    const float* bn_mean  = (const float*)d_in[25];
    const float* bn_var   = (const float*)d_in[26];
    const float* out_W    = (const float*)d_in[27];
    const float* out_b    = (const float*)d_in[28];
    float* out = (float*)d_out;

    // ---- workspace bump allocator ----
    char* wp = (char*)d_ws;
    auto alloc = [&](size_t bytes) -> void* {
        void* p = (void*)wp;
        wp += (bytes + 255) & ~(size_t)255;
        return p;
    };
    // f16 weight mirrors
    _Float16* w_msg  = (_Float16*)alloc(sizeof(_Float16) * REP * FIN);
    _Float16* w_lin  = (_Float16*)alloc(sizeof(_Float16) * REP * FIN);
    _Float16* w_lin2 = (_Float16*)alloc(sizeof(_Float16) * REP * 2 * REP);
    _Float16* w_g0ih = (_Float16*)alloc(sizeof(_Float16) * G3 * REP);
    _Float16* w_g0hh = (_Float16*)alloc(sizeof(_Float16) * G3 * REP);
    _Float16* w_g1ih = (_Float16*)alloc(sizeof(_Float16) * G3 * REP);
    _Float16* w_g1hh = (_Float16*)alloc(sizeof(_Float16) * G3 * REP);
    _Float16* w_cih  = (_Float16*)alloc(sizeof(_Float16) * G3 * REP);
    _Float16* w_chh  = (_Float16*)alloc(sizeof(_Float16) * G3 * REP);
    _Float16* w_hyp  = (_Float16*)alloc(sizeof(_Float16) * HYPD * REP);
    // activations
    _Float16* Xw16   = (_Float16*)alloc(sizeof(_Float16) * (size_t)NW * FIN);
    _Float16* Xt16   = (_Float16*)alloc(sizeof(_Float16) * (size_t)NW * FIN);
    float*    Psum   = (float*)   alloc(sizeof(float) * NW);
    float*    F32SCR = (float*)   alloc(sizeof(float) * (size_t)NW * REP);
    _Float16* H1_16  = (_Float16*)alloc(sizeof(_Float16) * (size_t)NW * REP);
    float*    XG     = (float*)   alloc(sizeof(float) * (size_t)NW * G3);   // reused: xg0, xg1, xg_cell
    _Float16* YS0    = (_Float16*)alloc(sizeof(_Float16) * (size_t)NW * REP); // GRU0 output (n,w,128)
    _Float16* YSN    = (_Float16*)alloc(sizeof(_Float16) * (size_t)NW * REP); // GRU1 output = h_nei
    _Float16* HS16   = (_Float16*)alloc(sizeof(_Float16) * (size_t)NW * REP); // h_self f16
    float*    HST    = (float*)   alloc(sizeof(float) * (size_t)NB * REP);    // GRU state f32
    _Float16* HST16  = (_Float16*)alloc(sizeof(_Float16) * (size_t)NB * REP);
    float*    HG     = (float*)   alloc(sizeof(float) * (size_t)NB * G3);     // h-gates / head out
    float*    U      = (float*)   alloc(sizeof(float) * (size_t)NB * REP);    // cell hx_mid f32
    _Float16* U16    = (_Float16*)alloc(sizeof(_Float16) * (size_t)NB * REP);
    float*    HXF    = (float*)   alloc(sizeof(float) * (size_t)NB * REP);    // cell hx f32
    _Float16* HX16   = (_Float16*)alloc(sizeof(_Float16) * (size_t)NB * REP);

    auto cvt = [&](const float* s, _Float16* d, int n) {
        hipLaunchKernelGGL(k_cvt_f16, dim3((n + 255) / 256), dim3(256), 0, stream, s, d, n);
    };
    auto zf32 = [&](float* p, int n) {
        hipLaunchKernelGGL(k_zero_f32, dim3((n + 255) / 256), dim3(256), 0, stream, p, n);
    };
    auto zf16 = [&](_Float16* p, int n) {
        hipLaunchKernelGGL(k_zero_f16, dim3((n + 255) / 256), dim3(256), 0, stream, p, n);
    };

    // ---- 0) f16 weight mirrors ----
    cvt(msg_W,  w_msg,  REP * FIN);
    cvt(lin_W,  w_lin,  REP * FIN);
    cvt(lin2_W, w_lin2, REP * 2 * REP);
    cvt(g0_Wih, w_g0ih, G3 * REP);
    cvt(g0_Whh, w_g0hh, G3 * REP);
    cvt(g1_Wih, w_g1ih, G3 * REP);
    cvt(g1_Whh, w_g1hh, G3 * REP);
    cvt(c_Wih,  w_cih,  G3 * REP);
    cvt(c_Whh,  w_chh,  G3 * REP);
    cvt(hyp_W,  w_hyp,  HYPD * REP);

    // ---- 1) fused neighbor reduce (single streaming pass over X) ----
    hipLaunchKernelGGL(k_reduce_neighbors, dim3(NW), dim3(FIN), 0, stream,
                       X, C, Xw16, Xt16, Psum);

    // ---- 2) h1 = msg_W @ Xw + Psum*msg_b   (28672 x 128 <- 256) ----
    launch_gemm(stream, Xw16, FIN, w_msg, FIN, F32SCR, REP, H1_16, REP,
                msg_b, Psum, NW, FIN, REP, 0, 0);

    // ---- 3) xg0 = h1 @ g0_Wih^T + bih   (28672 x 384 <- 128) ----
    launch_gemm(stream, H1_16, REP, w_g0ih, REP, XG, G3, nullptr, 0,
                g0_bih, nullptr, NW, REP, G3, 0, 0);

    // ---- 4) GRU layer 0 scan over W ----
    zf32(HST, NB * REP); zf16(HST16, NB * REP);
    for (int t = 0; t < WW; ++t) {
        launch_gemm(stream, HST16, REP, w_g0hh, REP, HG, G3, nullptr, 0,
                    g0_bhh, nullptr, NB, REP, G3, 0, 0);
        hipLaunchKernelGGL(k_gru_gates, dim3((NB * REP + 255) / 256), dim3(256), 0, stream,
                           XG + (size_t)t * G3, WW * G3, HG, HST,
                           HST, HST16, YS0 + (size_t)t * REP, WW * REP, NB);
    }

    // ---- 5) xg1 = ys0 @ g1_Wih^T + bih ----
    launch_gemm(stream, YS0, REP, w_g1ih, REP, XG, G3, nullptr, 0,
                g1_bih, nullptr, NW, REP, G3, 0, 0);

    // ---- 6) GRU layer 1 scan -> h_nei (YSN) ----
    zf32(HST, NB * REP); zf16(HST16, NB * REP);
    for (int t = 0; t < WW; ++t) {
        launch_gemm(stream, HST16, REP, w_g1hh, REP, HG, G3, nullptr, 0,
                    g1_bhh, nullptr, NB, REP, G3, 0, 0);
        hipLaunchKernelGGL(k_gru_gates, dim3((NB * REP + 255) / 256), dim3(256), 0, stream,
                           XG + (size_t)t * G3, WW * G3, HG, HST,
                           HST, HST16, YSN + (size_t)t * REP, WW * REP, NB);
    }

    // ---- 7) h_self = target_X @ lin_W^T + lin_b ----
    launch_gemm(stream, Xt16, FIN, w_lin, FIN, F32SCR, REP, HS16, REP,
                lin_b, nullptr, NW, FIN, REP, 0, 0);

    // ---- 8) xg_cell = h_self @ c_Wih^T + c_bih ----
    launch_gemm(stream, HS16, REP, w_cih, REP, XG, G3, nullptr, 0,
                c_bih, nullptr, NW, REP, G3, 0, 0);

    // ---- 9) cell scan ----
    zf16(HX16, NB * REP);
    for (int t = 0; t < WW; ++t) {
        // u = hx @ L1^T                 (lin2 columns 0..127)
        launch_gemm(stream, HX16, REP, w_lin2, 2 * REP, U, REP, nullptr, 0,
                    nullptr, nullptr, NB, REP, REP, 0, 0);
        // u = tanh(u + hn_t @ L2^T + b) (lin2 columns 128..255) -> hx_mid
        launch_gemm(stream, YSN + (size_t)t * REP, WW * REP, w_lin2 + REP, 2 * REP,
                    U, REP, U16, REP, lin2_b, nullptr, NB, REP, REP, 1, 1);
        // hg = hx_mid @ c_Whh^T + c_bhh
        launch_gemm(stream, U16, REP, w_chh, REP, HG, G3, nullptr, 0,
                    c_bhh, nullptr, NB, REP, G3, 0, 0);
        // hx = gru_gates(xg_cell[:,t], hg, hx_mid)
        hipLaunchKernelGGL(k_gru_gates, dim3((NB * REP + 255) / 256), dim3(256), 0, stream,
                           XG + (size_t)t * G3, WW * G3, HG, U,
                           HXF, HX16, (_Float16*)nullptr, 0, NB);
    }

    // ---- 10) head: h0 = hx @ hyp_W^T + hyp_b ----
    launch_gemm(stream, HX16, REP, w_hyp, REP, HG, HYPD, nullptr, 0,
                hyp_b, nullptr, NB, REP, HYPD, 0, 0);

    // ---- 11) BN + ReLU + out proj + sigmoid + loss ----
    zf32(out, 1);
    hipLaunchKernelGGL(k_head, dim3(NB), dim3(HYPD), 0, stream,
                       HG, Y, bn_gamma, bn_beta, bn_mean, bn_var, out_W, out_b, out);
}
